// Trans_40037685133832
// MI455X (gfx1250) — matrix operand order repaired and checked by reading, not compiled, not run on hardware
//
#include <hip/hip_runtime.h>
#include <hip/hip_bf16.h>

typedef __attribute__((ext_vector_type(16))) __bf16 v16bf;
typedef __attribute__((ext_vector_type(8)))  __bf16 v8bf;
typedef __attribute__((ext_vector_type(2)))  __bf16 v2bf;
typedef __attribute__((ext_vector_type(8)))  float  v8f;
typedef __attribute__((ext_vector_type(4)))  float  v4f;

#define DEV __device__ __forceinline__
#define PXB 64            // pixels per block
#define HW  16384         // 128*128
#define NB  8             // batch

DEV v8f combine8(v4f lo, v4f hi) {
  return __builtin_shufflevector(lo, hi, 0, 1, 2, 3, 4, 5, 6, 7);
}
DEV v16bf combine16(v8bf lo, v8bf hi) {
  return __builtin_shufflevector(lo, hi, 0,1,2,3,4,5,6,7,8,9,10,11,12,13,14,15);
}

// ---------------- WMMA fragment helpers (bf16 16x16x32, f32 accum) ----------------

// A matrix 16x32 (MxK) from PRE-PACKED bf16 weights wp[row][stride] (pads zero-filled).
DEV v16bf load_a_pack(const __bf16* __restrict__ wp, int stride, int mbase, int kbase, int lane) {
  const __bf16* p = wp + (size_t)(mbase + (lane & 15)) * stride + kbase + ((lane >> 4) << 3);
  v8bf lo = *(const v8bf*)(p);
  v8bf hi = *(const v8bf*)(p + 16);
  return combine16(lo, hi);
}

// B matrix 32x16 (KxN) from LDS pixel-major tile src[pix][chPad] bf16.
DEV v16bf load_b_frag(const __bf16* src, int chPad, int pixBase, int kbase, int lane) {
  const __bf16* p = src + (size_t)(pixBase + (lane & 15)) * chPad + kbase + ((lane >> 4) << 3);
  v8bf lo = *(const v8bf*)(p);
  v8bf hi = *(const v8bf*)(p + 16);
  return combine16(lo, hi);
}

DEV v8f wmma_bf16(v16bf a, v16bf b, v8f c) {
  return __builtin_amdgcn_wmma_f32_16x16x32_bf16(false, a, false, b, (short)0, c, false, false);
}

DEV float gelu_exact(float x) {
  return 0.5f * x * (1.0f + erff(x * 0.70710678118654752440f));
}

template<int MACT>
DEV v8f load_chan8(const float* p, int mbase, int lane) {
  int chan0 = mbase + ((lane >> 4) << 3);
  int cb = ((MACT % 16) == 0) ? chan0 : ((chan0 + 8 <= MACT) ? chan0 : (MACT - 8));
  return combine8(*(const v4f*)(p + cb), *(const v4f*)(p + cb + 4));
}

// D layout: lane&15 = pixel, VGPR r -> chan = mbase + (lane>>4)*8 + r.
// post-ops: +bias -> +residual -> *scale+shift -> act; masked rows -> 0.
// DST16: 16-B v8bf stores into [pix][dstPad] layout (LDS or global). GDSTF: f32 strided.
template<int MACT, int ACT, bool BIAS, bool BN, bool RESID, bool DST16, bool GDSTF>
DEV void store_d(v8f acc, int mbase, int pixBase, int lane,
                 v8f bv, v8f sv, v8f tv,
                 const __bf16* resid, int residPad,
                 __bf16* dst16, int dstPad, float* dstF, int gStride) {
  constexpr bool mAll = (MACT % 16) == 0;
  int chan0 = mbase + ((lane >> 4) << 3);
  int pix = pixBase + (lane & 15);
  v8f v = acc;
  if (BIAS) v = v + bv;
  if (RESID) {
    int cb = mAll ? chan0 : ((chan0 + 8 <= MACT) ? chan0 : (MACT - 8));
    v8bf rv = *(const v8bf*)(resid + (size_t)pix * residPad + cb);
    v = v + __builtin_convertvector(rv, v8f);
  }
  if (BN) v = v * sv + tv;
  if (ACT == 1) {
#pragma unroll
    for (int r = 0; r < 8; ++r) v[r] = fmaxf(v[r], 0.0f);
  } else if (ACT == 2) {
#pragma unroll
    for (int r = 0; r < 8; ++r) v[r] = gelu_exact(v[r]);
  }
  if (!mAll) {
#pragma unroll
    for (int r = 0; r < 8; ++r) if (chan0 + r >= MACT) v[r] = 0.0f;
  }
  v8bf o = __builtin_convertvector(v, v8bf);
  if (DST16) *(v8bf*)(dst16 + (size_t)pix * dstPad + chan0) = o;
  if (GDSTF) {
#pragma unroll
    for (int r = 0; r < 8; ++r) {
      int chan = chan0 + r;
      bool ok = mAll ? true : (chan < MACT);
      if (ok) dstF[(size_t)chan * gStride + pix] = v[r];
    }
  }
}

// One wave: OUT[MACT x 16px] = act(bn(W*X + b [+resid])); KACT selects K-step count.
template<int MACT, int KACT, int ACT, bool BIAS, bool BN, bool RESID, bool DST16, bool GDSTF>
DEV void conv16(const __bf16* wp, int wStride, const __bf16* src, int srcPad,
                const float* bias, const float* scale, const float* shift,
                const __bf16* resid, int residPad,
                __bf16* dst16, int dstPad, float* dstF, int gStride,
                int pixBase, int lane) {
  constexpr int KS = (KACT + 31) / 32;
  constexpr int MT = (MACT + 15) / 16;
  v16bf bf[KS];
#pragma unroll
  for (int ks = 0; ks < KS; ++ks) bf[ks] = load_b_frag(src, srcPad, pixBase, ks * 32, lane);
#pragma unroll
  for (int mt = 0; mt < MT; ++mt) {
    v8f acc = {0.f,0.f,0.f,0.f,0.f,0.f,0.f,0.f};
#pragma unroll
    for (int ks = 0; ks < KS; ++ks) {
      v16bf a = load_a_pack(wp, wStride, mt * 16, ks * 32, lane);
      acc = wmma_bf16(a, bf[ks], acc);
    }
    v8f bv = {0,0,0,0,0,0,0,0}, sv = bv, tv = bv;
    if (BIAS) bv = load_chan8<MACT>(bias, mt * 16, lane);
    if (BN) { sv = load_chan8<MACT>(scale, mt * 16, lane); tv = load_chan8<MACT>(shift, mt * 16, lane); }
    store_d<MACT, ACT, BIAS, BN, RESID, DST16, GDSTF>(
        acc, mt * 16, pixBase, lane, bv, sv, tv, resid, residPad, dst16, dstPad, dstF, gStride);
  }
}

// Prepared 48x48 conv (shared q/v projections): A-frags + epilogue hoisted across layers.
struct PC48 {
  v16bf a[6];
  v8f bv[3], sv[3], tv[3];
};

DEV void prep48(PC48& pc, const __bf16* wp, const float* b, const float* s, const float* t, int lane) {
#pragma unroll
  for (int mt = 0; mt < 3; ++mt) {
    pc.a[2 * mt]     = load_a_pack(wp, 64, mt * 16, 0, lane);
    pc.a[2 * mt + 1] = load_a_pack(wp, 64, mt * 16, 32, lane);
    pc.bv[mt] = load_chan8<48>(b, mt * 16, lane);
    pc.sv[mt] = load_chan8<48>(s, mt * 16, lane);
    pc.tv[mt] = load_chan8<48>(t, mt * 16, lane);
  }
}

DEV void apply48(const PC48& pc, const __bf16* src, __bf16* dst, int pixBase, int lane) {
  v16bf b0 = load_b_frag(src, 64, pixBase, 0, lane);
  v16bf b1 = load_b_frag(src, 64, pixBase, 32, lane);
#pragma unroll
  for (int mt = 0; mt < 3; ++mt) {
    v8f acc = {0.f,0.f,0.f,0.f,0.f,0.f,0.f,0.f};
    acc = wmma_bf16(pc.a[2 * mt], b0, acc);
    acc = wmma_bf16(pc.a[2 * mt + 1], b1, acc);
    store_d<48, 0, true, true, false, true, false>(
        acc, mt * 16, pixBase, lane, pc.bv[mt], pc.sv[mt], pc.tv[mt],
        nullptr, 0, dst, 64, nullptr, 0);
  }
}

// ---------------- weight packing: f32 row-major -> bf16 fragment-ready ----------------

__global__ __launch_bounds__(256) void pack_w(
    const float* __restrict__ W, int rowStride, int Kvalid, int M,
    __bf16* __restrict__ out, int outStride, int outCols, int colOff, int Mpad) {
  int i = blockIdx.x * 256 + threadIdx.x;
  if (i >= Mpad * outCols) return;
  int m = i / outCols, kk = i % outCols;
  __bf16 v = (__bf16)0.0f;
  if (m < M && kk < Kvalid) v = (__bf16)W[(size_t)m * rowStride + kk];
  out[(size_t)m * outStride + colOff + kk] = v;
}

// ---------------- adaptation: 48ch conv + BN ----------------
// TOUPS: write bf16 pixel-major [pix][64] (pads zeroed). Else: f32 channel-major tmp.

template<int CIN, int CINP, bool TOUPS>
__global__ __launch_bounds__(128) void adapt_kernel(
    const float* __restrict__ x, const __bf16* __restrict__ wp, int wStride,
    const float* __restrict__ s, const float* __restrict__ t,
    float* __restrict__ outF, __bf16* __restrict__ outB, int hw) {
  __shared__ __align__(16) __bf16 xs[PXB][CINP];
  int tid = threadIdx.x, lane = tid & 31, wave = tid >> 5;
  int pixBase = wave * 16;
  size_t gp0 = (size_t)blockIdx.x * PXB;
  int b = (int)(gp0 / hw), p0 = (int)(gp0 % hw);

  if (CINP > CIN) {
    for (int idx = tid; idx < PXB * CINP / 2; idx += 128)
      ((int*)&xs[0][0])[idx] = 0;
    __syncthreads();
  }
  for (int idx = tid; idx < CIN / 2 * PXB; idx += 128) {
    int c2 = (idx / PXB) * 2, px = idx % PXB;
    v2bf o;
    o[0] = (__bf16)x[((size_t)b * CIN + c2) * hw + p0 + px];
    o[1] = (__bf16)x[((size_t)b * CIN + c2 + 1) * hw + p0 + px];
    *(v2bf*)&xs[px][c2] = o;
  }
  __syncthreads();

  if (TOUPS) {
    __bf16* gout = outB + gp0 * 64;
    conv16<48, CIN, 0, false, true, false, true, false>(
        wp, wStride, &xs[0][0], CINP, nullptr, s, t, nullptr, 0,
        gout, 64, nullptr, 0, pixBase, lane);
    // zero the pad channels 48..63 of each pixel
    v8bf zz;
#pragma unroll
    for (int r = 0; r < 8; ++r) zz[r] = (__bf16)0.0f;
    *(v8bf*)(gout + (size_t)(pixBase + (lane & 15)) * 64 + 48 + ((lane >> 4) << 3)) = zz;
  } else {
    float* gout = outF + (size_t)b * 48 * hw + p0;
    conv16<48, CIN, 0, false, true, false, false, true>(
        wp, wStride, &xs[0][0], CINP, nullptr, s, t, nullptr, 0,
        nullptr, 0, gout, hw, pixBase, lane);
  }
}

// ---------------- bilinear upsample (half-pixel, edge clamp) -> bf16 padded ----------------

__global__ __launch_bounds__(256) void upsample_kernel(
    const float* __restrict__ in, __bf16* __restrict__ out, int hin) {
  size_t i = (size_t)blockIdx.x * 256 + threadIdx.x;
  const size_t total = (size_t)NB * HW * 32;   // 32 channel-pairs per pixel (incl pads)
  if (i >= total) return;
  int c2 = (int)(i & 31) * 2;
  size_t pixg = i >> 5;                        // b*HW + p
  v2bf o;
  o[0] = (__bf16)0.0f; o[1] = (__bf16)0.0f;
  if (c2 < 48) {
    int p = (int)(pixg & (HW - 1));
    int bb = (int)(pixg >> 14);
    int ox = p & 127, oy = p >> 7;
    float scale = (float)hin * (1.0f / 128.0f);
    float sx = (ox + 0.5f) * scale - 0.5f;
    float sy = (oy + 0.5f) * scale - 0.5f;
    int x0 = (int)floorf(sx), y0 = (int)floorf(sy);
    float fx = sx - (float)x0, fy = sy - (float)y0;
    int x1 = x0 + 1, y1 = y0 + 1;
    x0 = min(max(x0, 0), hin - 1); x1 = min(max(x1, 0), hin - 1);
    y0 = min(max(y0, 0), hin - 1); y1 = min(max(y1, 0), hin - 1);
    float w00 = (1.0f - fy) * (1.0f - fx), w01 = (1.0f - fy) * fx;
    float w10 = fy * (1.0f - fx),          w11 = fy * fx;
    size_t cs = (size_t)hin * hin;
    const float* pa = in + ((size_t)bb * 48 + c2) * cs;
    const float* pb = pa + cs;
    o[0] = (__bf16)(w00 * pa[y0 * hin + x0] + w01 * pa[y0 * hin + x1] +
                    w10 * pa[y1 * hin + x0] + w11 * pa[y1 * hin + x1]);
    o[1] = (__bf16)(w00 * pb[y0 * hin + x0] + w01 * pb[y0 * hin + x1] +
                    w10 * pb[y1 * hin + x0] + w11 * pb[y1 * hin + x1]);
  }
  *(v2bf*)(out + pixg * 64 + c2) = o;
}

// ---------------- fused 3-iteration attention pipeline ----------------

struct FusedParams {
  const __bf16 *ups0, *ups1, *ups2, *ups3;   // [NB*HW][64] bf16, pads zeroed
  const __bf16 *pw_rb1, *pw_rb2, *pw_q, *pw_v, *pw_k1, *pw_k2, *pw_cf, *pw_e1, *pw_e2;
  const float *rb_b1, *rb_s, *rb_t, *rb_b2, *bn_s, *bn_t;
  const float *q_b, *q_s, *q_t, *v_b, *v_s, *v_t;
  const float *k1_b, *k1_s, *k1_t, *k2_b, *k2_s, *k2_t;
  const float *cf_b, *cf_s, *cf_t;
  const float *emb_b1, *emb_b2;
  float* out;
};

__global__ __launch_bounds__(128) void fused_kernel(FusedParams P) {
  __shared__ __align__(16) __bf16 uT[4][PXB][64];    // async-staged, pads arrive zeroed
  __shared__ __align__(16) __bf16 qbuf[PXB][64];     // z / Q_l / V_l / refs
  __shared__ __align__(16) __bf16 labuf[PXB][64];    // la_res tile
  __shared__ __align__(16) __bf16 h24[PXB][32];      // 24-ch hidden (pads via masked store)
  __shared__ __align__(16) __bf16 kkbuf[2][PXB][32]; // keys (pads via masked store)

  int tid = threadIdx.x, lane = tid & 31, wave = tid >> 5;
  int pixBase = wave * 16;
  size_t gp0 = (size_t)blockIdx.x * PXB;
  int b = (int)(gp0 / HW), p0 = (int)(gp0 % HW);

  // async-stage the 4 layer tiles (8KB each) from bf16 pixel-major global into LDS
#pragma unroll
  for (int l = 0; l < 4; ++l) {
    const __bf16* u = (l == 0) ? P.ups0 : (l == 1) ? P.ups1 : (l == 2) ? P.ups2 : P.ups3;
    const char* gbase = (const char*)(u + gp0 * 64);
    char* lbase = (char*)&uT[l][0][0];
#pragma unroll
    for (int r = 0; r < 4; ++r) {
      int off = (tid + r * 128) * 16;
      unsigned long long ga = (unsigned long long)(gbase + off);
      unsigned int la = (unsigned int)(size_t)(lbase + off);
      asm volatile("global_load_async_to_lds_b128 %0, %1, off"
                   :: "v"(la), "v"(ga) : "memory");
    }
  }
  // zero pads of qbuf/labuf (other buffers get pads from masked stores / async copy)
  { int* z = (int*)&qbuf[0][0];  for (int i = tid; i < PXB*64/2; i += 128) z[i] = 0; }
  { int* z = (int*)&labuf[0][0]; for (int i = tid; i < PXB*64/2; i += 128) z[i] = 0; }
  asm volatile("s_wait_asynccnt 0x0" ::: "memory");
  __syncthreads();

  int mypix = tid >> 1, myj = tid & 1;

  for (int k = 0; k < 3; ++k) {
    // 1) res-block update: u = relu(bn(u + W2*relu(bn(W1*u+b1))+b2))
#pragma unroll
    for (int l = 0; l < 4; ++l) {
      __bf16* u = &uT[l][0][0];
      conv16<24, 48, 1, true, true, false, true, false>(
          P.pw_rb1, 64, u, 64, P.rb_b1, P.rb_s, P.rb_t, nullptr, 0,
          &h24[0][0], 32, nullptr, 0, pixBase, lane);
      __syncthreads();
      conv16<48, 24, 1, true, true, true, true, false>(
          P.pw_rb2, 32, &h24[0][0], 32, P.rb_b2, P.bn_s, P.bn_t, u, 64,
          u, 64, nullptr, 0, pixBase, lane);
      __syncthreads();
    }

    // 2) k==0: la = emb(relu(bn(cf * cat(ups))))
    if (k == 0) {
#pragma unroll
      for (int mt = 0; mt < 3; ++mt) {
        v8f acc = {0.f,0.f,0.f,0.f,0.f,0.f,0.f,0.f};
#pragma unroll
        for (int li = 0; li < 4; ++li) {
          const __bf16* src = &uT[li][0][0];
          v16bf bb0 = load_b_frag(src, 64, pixBase, 0, lane);
          v16bf bb1 = load_b_frag(src, 64, pixBase, 32, lane);
          v16bf aa0 = load_a_pack(P.pw_cf, 256, mt * 16, li * 64, lane);
          v16bf aa1 = load_a_pack(P.pw_cf, 256, mt * 16, li * 64 + 32, lane);
          acc = wmma_bf16(aa0, bb0, acc);
          acc = wmma_bf16(aa1, bb1, acc);
        }
        v8f bv = load_chan8<48>(P.cf_b, mt * 16, lane);
        v8f sv = load_chan8<48>(P.cf_s, mt * 16, lane);
        v8f tv = load_chan8<48>(P.cf_t, mt * 16, lane);
        store_d<48, 1, true, true, false, true, false>(
            acc, mt * 16, pixBase, lane, bv, sv, tv, nullptr, 0,
            &qbuf[0][0], 64, nullptr, 0);
      }
      __syncthreads();
      conv16<24, 48, 2, true, false, false, true, false>(
          P.pw_e1, 64, &qbuf[0][0], 64, P.emb_b1, nullptr, nullptr, nullptr, 0,
          &h24[0][0], 32, nullptr, 0, pixBase, lane);
      __syncthreads();
      conv16<48, 24, 0, true, false, false, true, false>(
          P.pw_e2, 32, &h24[0][0], 32, P.emb_b2, nullptr, nullptr, nullptr, 0,
          &labuf[0][0], 64, nullptr, 0, pixBase, lane);
      __syncthreads();
    }

    // 3) keys from la
    conv16<24, 48, 0, true, true, false, true, false>(
        P.pw_k1, 64, &labuf[0][0], 64, P.k1_b, P.k1_s, P.k1_t, nullptr, 0,
        &kkbuf[0][0][0], 32, nullptr, 0, pixBase, lane);
    conv16<24, 48, 0, true, true, false, true, false>(
        P.pw_k2, 64, &labuf[0][0], 64, P.k2_b, P.k2_s, P.k2_t, nullptr, 0,
        &kkbuf[1][0][0], 32, nullptr, 0, pixBase, lane);
    __syncthreads();

    // 4) Q per layer -> logits (thread owns (pixel, key); vector LDS reads)
    float logit[8];
    {
      PC48 qc;
      prep48(qc, P.pw_q, P.q_b, P.q_s, P.q_t, lane);
      v8f kkv[3];
#pragma unroll
      for (int c8 = 0; c8 < 3; ++c8)
        kkv[c8] = __builtin_convertvector(*(const v8bf*)&kkbuf[myj][mypix][c8 * 8], v8f);
#pragma unroll
      for (int l = 0; l < 4; ++l) {
        apply48(qc, &uT[l][0][0], &qbuf[0][0], pixBase, lane);
        __syncthreads();
#pragma unroll
        for (int h = 0; h < 2; ++h) {
          v8f s8 = {0.f,0.f,0.f,0.f,0.f,0.f,0.f,0.f};
#pragma unroll
          for (int c8 = 0; c8 < 3; ++c8) {
            v8f qv = __builtin_convertvector(*(const v8bf*)&qbuf[mypix][h * 24 + c8 * 8], v8f);
            s8 = s8 + kkv[c8] * qv;
          }
          float s = 0.0f;
#pragma unroll
          for (int r = 0; r < 8; ++r) s += s8[r];
          logit[l * 2 + h] = s;
        }
        __syncthreads();
      }
    }

    // softmax over 8 heads (registers)
    float att[8];
    float mx = logit[0];
#pragma unroll
    for (int e = 1; e < 8; ++e) mx = fmaxf(mx, logit[e]);
    float den = 0.0f;
#pragma unroll
    for (int e = 0; e < 8; ++e) { att[e] = expf(logit[e] - mx); den += att[e]; }
    float inv = 1.0f / den;
#pragma unroll
    for (int e = 0; e < 8; ++e) att[e] *= inv;

    // 5) V per layer -> ref accumulation (vector registers)
    v8f ref8[3];
#pragma unroll
    for (int c8 = 0; c8 < 3; ++c8) ref8[c8] = (v8f){0.f,0.f,0.f,0.f,0.f,0.f,0.f,0.f};
    {
      PC48 vc;
      prep48(vc, P.pw_v, P.v_b, P.v_s, P.v_t, lane);
#pragma unroll
      for (int l = 0; l < 4; ++l) {
        apply48(vc, &uT[l][0][0], &qbuf[0][0], pixBase, lane);
        __syncthreads();
#pragma unroll
        for (int c8 = 0; c8 < 3; ++c8) {
          v8f v0 = __builtin_convertvector(*(const v8bf*)&qbuf[mypix][c8 * 8], v8f);
          v8f v1 = __builtin_convertvector(*(const v8bf*)&qbuf[mypix][24 + c8 * 8], v8f);
          ref8[c8] = ref8[c8] + att[2 * l] * v0 + att[2 * l + 1] * v1;
        }
        __syncthreads();
      }
    }

    // write refs (concat over keys -> 48 channels) into qbuf
#pragma unroll
    for (int c8 = 0; c8 < 3; ++c8)
      *(v8bf*)&qbuf[mypix][myj * 24 + c8 * 8] = __builtin_convertvector(ref8[c8], v8bf);
    __syncthreads();

    // 6) la = emb(refs); final iteration writes straight to global output
    conv16<24, 48, 2, true, false, false, true, false>(
        P.pw_e1, 64, &qbuf[0][0], 64, P.emb_b1, nullptr, nullptr, nullptr, 0,
        &h24[0][0], 32, nullptr, 0, pixBase, lane);
    __syncthreads();
    if (k < 2) {
      conv16<48, 24, 0, true, false, false, true, false>(
          P.pw_e2, 32, &h24[0][0], 32, P.emb_b2, nullptr, nullptr, nullptr, 0,
          &labuf[0][0], 64, nullptr, 0, pixBase, lane);
    } else {
      float* gout = P.out + (size_t)b * 48 * HW + p0;
      conv16<48, 24, 0, true, false, false, false, true>(
          P.pw_e2, 32, &h24[0][0], 32, P.emb_b2, nullptr, nullptr, nullptr, 0,
          nullptr, 0, gout, HW, pixBase, lane);
    }
    __syncthreads();
  }
}

// ---------------- host launcher ----------------

extern "C" void kernel_launch(void* const* d_in, const int* in_sizes, int n_in,
                              void* d_out, int out_size, void* d_ws, size_t ws_size,
                              hipStream_t stream) {
  (void)in_sizes; (void)n_in; (void)out_size; (void)ws_size;
  #define IN(i) ((const float*)d_in[(i)])

  const float* x0 = IN(0);  const float* x1 = IN(1);
  const float* x2 = IN(2);  const float* x3 = IN(3);

  char* base = (char*)d_ws;
  const size_t UPSB = (size_t)NB * HW * 64 * sizeof(__bf16);   // 16 MB each
  __bf16* ups0 = (__bf16*)(base);
  __bf16* ups1 = (__bf16*)(base + UPSB);
  __bf16* ups2 = (__bf16*)(base + 2 * UPSB);
  __bf16* ups3 = (__bf16*)(base + 3 * UPSB);
  float* tmp1 = (float*)(base + 4 * UPSB);                      // 8*48*4096 f32
  float* tmp2 = tmp1 + (size_t)NB * 48 * 4096;                  // 8*48*1024 f32
  float* tmp3 = tmp2 + (size_t)NB * 48 * 1024;                  // 8*48*256  f32

  // packed bf16 weights region (after all other buffers; 16B aligned)
  __bf16* wb = (__bf16*)(tmp3 + (size_t)NB * 48 * 256);
  __bf16* pw_rb1 = wb;             // 32 x 64
  __bf16* pw_rb2 = pw_rb1 + 2048;  // 48 x 32
  __bf16* pw_q   = pw_rb2 + 1536;  // 48 x 64
  __bf16* pw_v   = pw_q   + 3072;  // 48 x 64
  __bf16* pw_k1  = pw_v   + 3072;  // 32 x 64
  __bf16* pw_k2  = pw_k1  + 2048;  // 32 x 64
  __bf16* pw_cf  = pw_k2  + 2048;  // 48 x 256 (4 segments of 64)
  __bf16* pw_e1  = pw_cf  + 12288; // 32 x 64
  __bf16* pw_e2  = pw_e1  + 2048;  // 48 x 32
  __bf16* pw_a0  = pw_e2  + 1536;  // 48 x 64
  __bf16* pw_a1  = pw_a0  + 3072;  // 48 x 96
  __bf16* pw_a2  = pw_a1  + 4608;  // 48 x 192
  __bf16* pw_a3  = pw_a2  + 9216;  // 48 x 384

  #define PACK(W, rs, Kv, M, out, ostr, ocols, coff, Mp) \
    pack_w<<<((Mp) * (ocols) + 255) / 256, 256, 0, stream>>>(W, rs, Kv, M, out, ostr, ocols, coff, Mp)

  PACK(IN(40), 48, 48, 24, pw_rb1, 64, 64, 0, 32);    // rb_w1
  PACK(IN(44), 24, 24, 48, pw_rb2, 32, 32, 0, 48);    // rb_w2
  PACK(IN(24), 48, 48, 48, pw_q,   64, 64, 0, 48);    // q_w
  PACK(IN(28), 48, 48, 48, pw_v,   64, 64, 0, 48);    // v_w
  PACK(IN(16), 48, 48, 24, pw_k1,  64, 64, 0, 32);    // k1_w
  PACK(IN(20), 48, 48, 24, pw_k2,  64, 64, 0, 32);    // k2_w
  for (int li = 0; li < 4; ++li)                      // cf_w segments
    PACK(IN(32) + 48 * li, 192, 48, 48, pw_cf, 256, 64, 64 * li, 48);
  PACK(IN(36), 48, 48, 24, pw_e1,  64, 64, 0, 32);    // emb_w1
  PACK(IN(38), 24, 24, 48, pw_e2,  32, 32, 0, 48);    // emb_w2
  PACK(IN(4),  48, 48, 48, pw_a0,  64, 64, 0, 48);    // adap_w0
  PACK(IN(7),  96, 96, 48, pw_a1,  96, 96, 0, 48);    // adap_w1
  PACK(IN(10), 192, 192, 48, pw_a2, 192, 192, 0, 48); // adap_w2
  PACK(IN(13), 384, 384, 48, pw_a3, 384, 384, 0, 48); // adap_w3
  #undef PACK

  adapt_kernel<48, 64, true><<<NB * HW / PXB, 128, 0, stream>>>(
      x0, pw_a0, 64, IN(5), IN(6), nullptr, ups0, HW);
  adapt_kernel<96, 96, false><<<NB * 4096 / PXB, 128, 0, stream>>>(
      x1, pw_a1, 96, IN(8), IN(9), tmp1, nullptr, 4096);
  adapt_kernel<192, 192, false><<<NB * 1024 / PXB, 128, 0, stream>>>(
      x2, pw_a2, 192, IN(11), IN(12), tmp2, nullptr, 1024);
  adapt_kernel<384, 384, false><<<NB * 256 / PXB, 128, 0, stream>>>(
      x3, pw_a3, 384, IN(14), IN(15), tmp3, nullptr, 256);

  const int upsBlocks = (int)(((size_t)NB * HW * 32 + 255) / 256);
  upsample_kernel<<<upsBlocks, 256, 0, stream>>>(tmp1, ups1, 64);
  upsample_kernel<<<upsBlocks, 256, 0, stream>>>(tmp2, ups2, 32);
  upsample_kernel<<<upsBlocks, 256, 0, stream>>>(tmp3, ups3, 16);

  FusedParams P;
  P.ups0 = ups0; P.ups1 = ups1; P.ups2 = ups2; P.ups3 = ups3;
  P.pw_rb1 = pw_rb1; P.pw_rb2 = pw_rb2; P.pw_q = pw_q; P.pw_v = pw_v;
  P.pw_k1 = pw_k1; P.pw_k2 = pw_k2; P.pw_cf = pw_cf; P.pw_e1 = pw_e1; P.pw_e2 = pw_e2;
  P.k1_b = IN(17); P.k1_s = IN(18); P.k1_t = IN(19);
  P.k2_b = IN(21); P.k2_s = IN(22); P.k2_t = IN(23);
  P.q_b  = IN(25); P.q_s  = IN(26); P.q_t  = IN(27);
  P.v_b  = IN(29); P.v_s  = IN(30); P.v_t  = IN(31);
  P.cf_b = IN(33); P.cf_s = IN(34); P.cf_t = IN(35);
  P.emb_b1 = IN(37); P.emb_b2 = IN(39);
  P.rb_b1 = IN(41); P.rb_s = IN(42); P.rb_t = IN(43);
  P.rb_b2 = IN(45);
  P.bn_s = IN(46); P.bn_t = IN(47);
  P.out = (float*)d_out;

  fused_kernel<<<NB * HW / PXB, 128, 0, stream>>>(P);
  #undef IN
}
